// Transformer_drop_86620900426088
// MI455X (gfx1250) — compile-verified, matrix operand order checked by reading, not run on hardware
//
#include <hip/hip_runtime.h>
#include <hip/hip_bf16.h>
#include <math.h>

#define HH 12
#define CC 768
#define BB 16
#define MPAD 3200   // 197*16 = 3152 rounded up to 64; all other layer Ms divide 64

typedef __attribute__((ext_vector_type(16))) _Float16 v16h;
typedef __attribute__((ext_vector_type(8)))  _Float16 v8h;
typedef __attribute__((ext_vector_type(8)))  float    v8f;
typedef __attribute__((ext_vector_type(4)))  unsigned int u32x4;
typedef __attribute__((ext_vector_type(8)))  int      i32x8;
typedef __attribute__((ext_vector_type(4)))  int      i32x4;

#if defined(__has_builtin)
#if __has_builtin(__builtin_amdgcn_tensor_load_to_lds) && __has_builtin(__builtin_amdgcn_s_wait_tensorcnt)
#define USE_TDM 1
#endif
#endif
#ifndef USE_TDM
#define USE_TDM 0
#endif

// ---------------------------------------------------------------------------
// Fragment helpers for V_WMMA_F32_16X16X32_F16 (wave32).
// Lane L<16 holds row base+L with K = k+0..7 / k+16..23; lanes 16-31 hold the
// same rows with K = k+8..15 / k+24..31. Two 16B chunks per lane per fragment.
// ---------------------------------------------------------------------------
__device__ __forceinline__ v16h frag_from_f16(const _Float16* base, int ld,
                                              int row, int k0, int k1) {
  const _Float16* p = base + (size_t)row * ld;
  v8h lo = *(const v8h*)(p + k0);
  v8h hi = *(const v8h*)(p + k1);
  v16h r;
#pragma unroll
  for (int i = 0; i < 8; ++i) { r[i] = lo[i]; r[i + 8] = hi[i]; }
  return r;
}

__device__ __forceinline__ v16h frag_from_f32(const float* __restrict__ base,
                                              int ld, int row, int k0, int k1) {
  const float* p = base + (size_t)row * ld;
  v16h r;
#pragma unroll
  for (int i = 0; i < 8; ++i) r[i] = (_Float16)p[k0 + i];
#pragma unroll
  for (int i = 0; i < 8; ++i) r[i + 8] = (_Float16)p[k1 + i];
  return r;
}

// ---------------------------------------------------------------------------
// Issue one TDM 2D tile load: 64 rows x 128 cols of f16 (16 KB) from global
// (row stride K elements) into LDS at byte offset ldsOff. D# built per
// CDNA5 ISA; type=2 ("image"), count=1,
// data_size=1 (2 bytes), tile_dim0=128, tile_dim1=64.
// 6-arg builtin form (clang-23 / therock-10.0 headers).
// ---------------------------------------------------------------------------
__device__ __forceinline__ void tdm_load_tile(const _Float16* gsrc, int K,
                                              unsigned ldsOff) {
#if USE_TDM
  const unsigned long long ga = (unsigned long long)(uintptr_t)gsrc;
  u32x4 g0;
  g0[0] = 1u;                                    // count=1, user descriptor
  g0[1] = ldsOff;                                // lds_addr
  g0[2] = (unsigned)(ga & 0xFFFFFFFFu);          // global_addr[31:0]
  g0[3] = (unsigned)((ga >> 32) & 0x1FFFFFFu)    // global_addr[56:32]
          | 0x80000000u;                         // type=2 at bits[127:126]
  i32x8 g1;
  g1[0] = 0x10000;                               // data_size=1 (2B)
  g1[1] = (K & 0xFFFF) << 16;                    // tensor_dim0[15:0]
  g1[2] = (K >> 16) | ((MPAD & 0xFFFF) << 16);   // tensor_dim0[31:16] | tensor_dim1[15:0]
  g1[3] = (MPAD >> 16) | (128 << 16);            // tensor_dim1[31:16] | tile_dim0=128
  g1[4] = 64;                                    // tile_dim1=64, tile_dim2=0
  g1[5] = K;                                     // tensor_dim0_stride[31:0]
  g1[6] = 0;                                     // stride hi, dim1_stride lo
  g1[7] = 0;
  i32x4 z4 = {};
  i32x8 z8 = {};
  __builtin_amdgcn_tensor_load_to_lds(g0, g1, z4, z4, z8, 0);
#else
  (void)gsrc; (void)K; (void)ldsOff;
#endif
}

// ---------------------------------------------------------------------------
// GEMM: Out[M,N] = A[M,K](f16) * W[N,K]^T(f32) + bias.
//   EPI 0: bias -> f32 ; EPI 1: bias + residual -> f32 ; EPI 2: bias+quickGELU -> f16
// Block = 256 threads = 8 waves (2x4); block tile 64x128; wave tile 32x32 via
// 2x2 v_wmma_f32_16x16x32_f16 fragments. A tile staged LDS-resident through
// the Tensor Data Mover in double-buffered 64x128 chunks; W converted f32->f16
// in-register. All buffers padded to MPAD rows -> unconditional stores.
// ---------------------------------------------------------------------------
template<int EPI>
__global__ __launch_bounds__(256)
void gemm_wmma_kernel(const _Float16* __restrict__ A, const float* __restrict__ W,
                      const float* __restrict__ bias, const float* __restrict__ R,
                      float* __restrict__ O32, _Float16* __restrict__ O16,
                      int M, int N, int K) {
  __shared__ _Float16 smem[2][64 * 128];         // 32 KB double buffer
  const int tid  = threadIdx.x;
  const int lane = tid & 31;
  const int wave = tid >> 5;
  const int l16  = lane & 15;
  const int hi   = lane >> 4;

  const int mBase0 = blockIdx.y * 64;            // block row origin
  const int wRow   = (wave >> 2) * 32;           // wave row inside block tile
  const int nBase  = blockIdx.x * 128 + (wave & 3) * 32;

  const unsigned lds0 = (unsigned)(uintptr_t)&smem[0][0];
  const unsigned lds1 = (unsigned)(uintptr_t)&smem[1][0];
  const _Float16* gA = A + (size_t)mBase0 * K;

  v8f acc[2][2] = {};

  const int nch = K >> 7;                        // K / 128 chunks
#if USE_TDM
  if (wave == 0) {
    tdm_load_tile(gA, K, lds0);
    __builtin_amdgcn_s_wait_tensorcnt(0);
  }
  __syncthreads();
#else
  // Fallback: cooperative staging of chunk 0.
  {
    const int r = tid >> 2, c = (tid & 3) * 32;  // 64 rows x 128 cols, 16B units
#pragma unroll
    for (int j = 0; j < 4; ++j)
      *(v8h*)&smem[0][r * 128 + c + j * 8] = *(const v8h*)(gA + (size_t)r * K + c + j * 8);
  }
  __syncthreads();
#endif

  for (int ch = 0; ch < nch; ++ch) {
    const int cur = ch & 1;
#if USE_TDM
    if (wave == 0 && ch + 1 < nch)
      tdm_load_tile(gA + (ch + 1) * 128, K, cur ? lds0 : lds1);
#endif
    const _Float16* sa = &smem[cur][0];
    const int kcBase = ch * 128;
#pragma unroll
    for (int ks = 0; ks < 4; ++ks) {
      const int kk0 = ks * 32 + hi * 8;
      const int kk1 = ks * 32 + 16 + hi * 8;
      v16h a0 = frag_from_f16(sa, 128, wRow + l16,      kk0, kk1);
      v16h a1 = frag_from_f16(sa, 128, wRow + 16 + l16, kk0, kk1);
      v16h b0 = frag_from_f32(W, K, nBase + l16,      kcBase + kk0, kcBase + kk1);
      v16h b1 = frag_from_f32(W, K, nBase + 16 + l16, kcBase + kk0, kcBase + kk1);
      acc[0][0] = __builtin_amdgcn_wmma_f32_16x16x32_f16(false, a0, false, b0, (short)0, acc[0][0], false, false);
      acc[0][1] = __builtin_amdgcn_wmma_f32_16x16x32_f16(false, a0, false, b1, (short)0, acc[0][1], false, false);
      acc[1][0] = __builtin_amdgcn_wmma_f32_16x16x32_f16(false, a1, false, b0, (short)0, acc[1][0], false, false);
      acc[1][1] = __builtin_amdgcn_wmma_f32_16x16x32_f16(false, a1, false, b1, (short)0, acc[1][1], false, false);
    }
#if USE_TDM
    if (wave == 0) __builtin_amdgcn_s_wait_tensorcnt(0);
    __syncthreads();
#else
    __syncthreads();
    if (ch + 1 < nch) {
      const int r = tid >> 2, c = (tid & 3) * 32;
      const _Float16* gsrc = gA + (ch + 1) * 128;
#pragma unroll
      for (int j = 0; j < 4; ++j)
        *(v8h*)&smem[1 - cur][r * 128 + c + j * 8] = *(const v8h*)(gsrc + (size_t)r * K + c + j * 8);
    }
    __syncthreads();
#endif
  }
  (void)M;

  // C/D layout: VGPR r, lanes 0-15 -> row r, lanes 16-31 -> row r+8; col = lane&15.
  const int mW = mBase0 + wRow;
#pragma unroll
  for (int mi = 0; mi < 2; ++mi)
#pragma unroll
    for (int ni = 0; ni < 2; ++ni)
#pragma unroll
      for (int r = 0; r < 8; ++r) {
        const int row = mW + mi * 16 + hi * 8 + r;
        const int col = nBase + ni * 16 + l16;
        float v = acc[mi][ni][r] + bias[col];
        if (EPI == 1) v += R[(size_t)row * N + col];
        if (EPI == 2) v = v * (1.0f / (1.0f + __expf(-1.702f * v)));   // quick_gelu
        if (EPI == 2) O16[(size_t)row * N + col] = (_Float16)v;
        else          O32[(size_t)row * N + col] = v;
      }
}

// ---------------------------------------------------------------------------
// LayerNorm over C=768, one block per token row, f16 output (WMMA A feed).
// ---------------------------------------------------------------------------
__global__ __launch_bounds__(256)
void ln_f16_kernel(const float* __restrict__ x, const float* __restrict__ g,
                   const float* __restrict__ b, _Float16* __restrict__ y, int rows) {
  __shared__ float red[256];
  const int row = blockIdx.x;
  const int tid = threadIdx.x;
  if (row >= rows) return;
  const float* xr = x + (size_t)row * CC;
  float v0 = xr[tid], v1 = xr[tid + 256], v2 = xr[tid + 512];
  red[tid] = v0 + v1 + v2;
  __syncthreads();
  for (int off = 128; off; off >>= 1) { if (tid < off) red[tid] += red[tid + off]; __syncthreads(); }
  const float mean = red[0] * (1.0f / CC);
  __syncthreads();
  const float d0 = v0 - mean, d1 = v1 - mean, d2 = v2 - mean;
  red[tid] = d0 * d0 + d1 * d1 + d2 * d2;
  __syncthreads();
  for (int off = 128; off; off >>= 1) { if (tid < off) red[tid] += red[tid + off]; __syncthreads(); }
  const float rstd = rsqrtf(red[0] * (1.0f / CC) + 1e-5f);
  _Float16* yr = y + (size_t)row * CC;
  yr[tid]       = (_Float16)(d0 * rstd * g[tid]       + b[tid]);
  yr[tid + 256] = (_Float16)(d1 * rstd * g[tid + 256] + b[tid + 256]);
  yr[tid + 512] = (_Float16)(d2 * rstd * g[tid + 512] + b[tid + 512]);
}

// ---------------------------------------------------------------------------
// Attention: one block per (b,h). Row-wise softmax(QK^T/8)V; writes o as f16
// for the proj GEMM and the CLS attention row (for token dropping).
// ---------------------------------------------------------------------------
__global__ __launch_bounds__(256)
void attn_kernel(const float* __restrict__ qkv, _Float16* __restrict__ o16,
                 float* __restrict__ cls, int N) {
  const int bh = blockIdx.x;
  const int b = bh / HH, h = bh % HH;
  const int tid = threadIdx.x;
  __shared__ float qs[64];
  __shared__ float s[224];
  __shared__ float red[256];
  __shared__ float ssum;
  const size_t st = 3 * CC;
  const int hoff = h * 64;

  for (int n = 0; n < N; ++n) {
    if (tid < 64) qs[tid] = qkv[((size_t)n * BB + b) * st + hoff + tid];
    __syncthreads();
    for (int m = tid; m < N; m += 256) {
      const float* kr = qkv + ((size_t)m * BB + b) * st + CC + hoff;
      float acc = 0.f;
#pragma unroll
      for (int d = 0; d < 64; ++d) acc += qs[d] * kr[d];
      s[m] = acc * 0.125f;   // 1/sqrt(64)
    }
    __syncthreads();
    red[tid] = (tid < N) ? s[tid] : -1e30f;
    __syncthreads();
    for (int off = 128; off; off >>= 1) { if (tid < off) red[tid] = fmaxf(red[tid], red[tid + off]); __syncthreads(); }
    const float mx = red[0];
    __syncthreads();
    float pv = 0.f;
    if (tid < N) { pv = __expf(s[tid] - mx); s[tid] = pv; }
    red[tid] = pv;
    __syncthreads();
    for (int off = 128; off; off >>= 1) { if (tid < off) red[tid] += red[tid + off]; __syncthreads(); }
    if (tid == 0) ssum = red[0];
    __syncthreads();
    const float inv = 1.0f / ssum;
    if (n == 0) {
      for (int m = tid; m < N; m += 256) cls[(size_t)bh * N + m] = s[m] * inv;
    }
    const int d = tid & 63, g = tid >> 6;
    float acc = 0.f;
    for (int m = g; m < N; m += 4)
      acc += s[m] * qkv[((size_t)m * BB + b) * st + 2 * CC + hoff + d];
    red[tid] = acc;
    __syncthreads();
    if (g == 0)
      o16[((size_t)n * BB + b) * CC + hoff + d] =
          (_Float16)((red[d] + red[64 + d] + red[128 + d] + red[192 + d]) * inv);
    __syncthreads();
  }
}

// ---------------------------------------------------------------------------
// EViT token drop: per batch, head-mean CLS attention, iterative argmax top-k
// (descending value, smaller index ties = jax top_k order), gather kept rows,
// build attention-weighted fused token.
// ---------------------------------------------------------------------------
__global__ __launch_bounds__(256)
void drop_kernel(const float* __restrict__ x, const float* __restrict__ cls,
                 float* __restrict__ xo, int N, int left) {
  const int b = blockIdx.x, tid = threadIdx.x;
  __shared__ float ca[200];
  __shared__ int   used[200];
  __shared__ float rv[256];
  __shared__ int   ri[256];
  __shared__ int   order[160];
  const int nk = N - 1;

  if (tid < nk) {
    float acc = 0.f;
    for (int h = 0; h < HH; ++h) acc += cls[((size_t)b * HH + h) * N + (tid + 1)];
    ca[tid] = acc * (1.0f / HH);
    used[tid] = 0;
  }
  __syncthreads();

  for (int k = 0; k < left; ++k) {
    rv[tid] = (tid < nk && !used[tid]) ? ca[tid] : -1e30f;
    ri[tid] = tid;
    __syncthreads();
    for (int off = 128; off; off >>= 1) {
      if (tid < off) {
        if (rv[tid + off] > rv[tid] ||
            (rv[tid + off] == rv[tid] && ri[tid + off] < ri[tid])) {
          rv[tid] = rv[tid + off];
          ri[tid] = ri[tid + off];
        }
      }
      __syncthreads();
    }
    if (tid == 0) { order[k] = ri[0]; used[ri[0]] = 1; }
    __syncthreads();
  }

  for (int c = tid; c < CC; c += 256)
    xo[(size_t)b * CC + c] = x[(size_t)b * CC + c];
  for (int k = 0; k < left; ++k) {
    const int src = order[k] + 1;
    for (int c = tid; c < CC; c += 256)
      xo[((size_t)(1 + k) * BB + b) * CC + c] = x[((size_t)src * BB + b) * CC + c];
  }
  for (int c = tid; c < CC; c += 256) {
    float acc = 0.f;
    for (int m = 0; m < nk; ++m)
      if (!used[m]) acc += ca[m] * x[((size_t)(m + 1) * BB + b) * CC + c];
    xo[((size_t)(left + 1) * BB + b) * CC + c] = acc;
  }
}

__global__ void copy_f32_kernel(const float* __restrict__ src, float* __restrict__ dst, int n) {
  const int i = blockIdx.x * 256 + threadIdx.x;
  if (i < n) dst[i] = src[i];
}

// ---------------------------------------------------------------------------
extern "C" void kernel_launch(void* const* d_in, const int* in_sizes, int n_in,
                              void* d_out, int out_size, void* d_ws, size_t ws_size,
                              hipStream_t stream) {
  (void)in_sizes; (void)n_in; (void)ws_size;
  const float* x_in  = (const float*)d_in[0];
  const float* w_in  = (const float*)d_in[1];
  const float* b_in  = (const float*)d_in[2];
  const float* w_out = (const float*)d_in[3];
  const float* b_out = (const float*)d_in[4];
  const float* ln1_g = (const float*)d_in[5];
  const float* ln1_b = (const float*)d_in[6];
  const float* w_fc  = (const float*)d_in[7];
  const float* b_fc  = (const float*)d_in[8];
  const float* w_pr  = (const float*)d_in[9];
  const float* b_pr  = (const float*)d_in[10];
  const float* ln2_g = (const float*)d_in[11];
  const float* ln2_b = (const float*)d_in[12];

  const int N0v = 197;

  char* ws = (char*)d_ws;
  size_t off = 0;
  auto alloc = [&](size_t bytes) -> void* {
    void* p = ws + off;
    off = (off + bytes + 255) & ~(size_t)255;
    return p;
  };
  // All M-dimension buffers padded to MPAD rows (unconditional GEMM stores).
  float*    xA   = (float*)   alloc((size_t)MPAD * CC * 4);
  float*    xB   = (float*)   alloc((size_t)MPAD * CC * 4);
  _Float16* ln16 = (_Float16*)alloc((size_t)MPAD * CC * 2);
  float*    qkv  = (float*)   alloc((size_t)MPAD * 3 * CC * 4);
  _Float16* o16  = (_Float16*)alloc((size_t)MPAD * CC * 2);
  _Float16* h16  = (_Float16*)alloc((size_t)MPAD * 4 * CC * 2);
  float*    cls  = (float*)   alloc((size_t)BB * HH * N0v * 4);

  (void)hipMemcpyAsync(xA, x_in, (size_t)N0v * BB * CC * 4, hipMemcpyDeviceToDevice, stream);

  float* cur = xA;
  float* alt = xB;
  int N = N0v;

  for (int l = 0; l < 12; ++l) {
    const int M = N * BB;
    const float* Wi = w_in  + (size_t)l * 3 * CC * CC;
    const float* Bi = b_in  + (size_t)l * 3 * CC;
    const float* Wo = w_out + (size_t)l * CC * CC;
    const float* Bo = b_out + (size_t)l * CC;
    const float* Wf = w_fc  + (size_t)l * 4 * CC * CC;
    const float* Bf = b_fc  + (size_t)l * 4 * CC;
    const float* Wp = w_pr  + (size_t)l * CC * 4 * CC;
    const float* Bp = b_pr  + (size_t)l * CC;

    // x_ln = LN1(x); qkv = x_ln @ Wi^T + Bi
    ln_f16_kernel<<<M, 256, 0, stream>>>(cur, ln1_g + l * CC, ln1_b + l * CC, ln16, M);
    gemm_wmma_kernel<0><<<dim3(3 * CC / 128, (M + 63) / 64), 256, 0, stream>>>(
        ln16, Wi, Bi, nullptr, qkv, nullptr, M, 3 * CC, CC);
    // o = softmax(QK^T/8)V ; cls attention row captured
    attn_kernel<<<BB * HH, 256, 0, stream>>>(qkv, o16, cls, N);
    // x = x + o @ Wo^T + Bo
    gemm_wmma_kernel<1><<<dim3(CC / 128, (M + 63) / 64), 256, 0, stream>>>(
        o16, Wo, Bo, cur, cur, nullptr, M, CC, CC);

    if (l == 3 || l == 6 || l == 9) {            // DPR = 0.3 layers
      const int left = (7 * (N - 1) + 9) / 10;   // ceil(0.7*(N-1))
      drop_kernel<<<BB, 256, 0, stream>>>(cur, cls, alt, N, left);
      float* t = cur; cur = alt; alt = t;
      N = left + 2;                              // cls + kept + fused
    }

    const int M2 = N * BB;
    // h = quickGELU(LN2(x) @ Wf^T + Bf); x = x + h @ Wp^T + Bp
    ln_f16_kernel<<<M2, 256, 0, stream>>>(cur, ln2_g + l * CC, ln2_b + l * CC, ln16, M2);
    gemm_wmma_kernel<2><<<dim3(4 * CC / 128, (M2 + 63) / 64), 256, 0, stream>>>(
        ln16, Wf, Bf, nullptr, nullptr, h16, M2, 4 * CC, CC);
    gemm_wmma_kernel<1><<<dim3(CC / 128, (M2 + 63) / 64), 256, 0, stream>>>(
        h16, Wp, Bp, cur, cur, nullptr, M2, CC, 4 * CC);
  }

  copy_f32_kernel<<<(out_size + 255) / 256, 256, 0, stream>>>(cur, (float*)d_out, out_size);
}
